// TransIFFEncoder_60172491817903
// MI455X (gfx1250) — matrix operand order repaired and checked by reading, not compiled, not run on hardware
//
#include <hip/hip_runtime.h>
#include <hip/hip_bf16.h>
#include <math.h>

#define D    256
#define NH   8
#define HD   32
#define KK   5
#define K2   25
#define NLAYER 2
#define DFF  1024
#define EPS  1e-5f
#define BCH  8          // k-steps staged per LDS chunk (4 tiles * 8 steps * 1KB = 32KB)

typedef __attribute__((ext_vector_type(16))) __bf16 v16bf;
typedef __attribute__((ext_vector_type(8)))  __bf16 v8bf;
typedef __attribute__((ext_vector_type(8)))  float  v8f;
typedef __attribute__((ext_vector_type(4)))  unsigned int u32x4;
typedef __attribute__((ext_vector_type(8)))  int i32x8;
typedef __attribute__((ext_vector_type(4)))  int i32x4;

#ifndef __has_builtin
#define __has_builtin(x) 0
#endif

#if defined(__AMDGCN__) && __has_builtin(__builtin_amdgcn_tensor_load_to_lds)
#define USE_TDM 1
#endif

// ---------- helpers ----------
__device__ __forceinline__ __bf16 f2bf(float f) {
    unsigned u = __builtin_bit_cast(unsigned, f);
    unsigned r = u + 0x7FFFu + ((u >> 16) & 1u);   // round-to-nearest-even
    unsigned short s = (unsigned short)(r >> 16);
    return __builtin_bit_cast(__bf16, s);
}
__device__ __forceinline__ v16bf cat16(v8bf lo, v8bf hi) {
    return __builtin_shufflevector(lo, hi, 0,1,2,3,4,5,6,7,8,9,10,11,12,13,14,15);
}
__device__ __forceinline__ void wait_tensor0() {
#if defined(__AMDGCN__)
#if __has_builtin(__builtin_amdgcn_s_wait_tensorcnt)
    __builtin_amdgcn_s_wait_tensorcnt(0);
#else
    asm volatile("s_wait_tensorcnt 0x0" ::: "memory");
#endif
#endif
}

#if defined(USE_TDM)
// TDM 2D tile load: nRows rows of rowDw dwords, source row stride strideDw dwords,
// packed tight into LDS at lds_off.  D# per CDNA5 ISA.
__device__ __forceinline__ void tdm_load2d(unsigned lds_off, const void* gsrc,
                                           unsigned rowDw, unsigned nRows, unsigned strideDw)
{
    unsigned long long ga = (unsigned long long)(size_t)gsrc;
    u32x4 g0;
    g0.x = 1u;                                                   // count=1 (valid), user mode
    g0.y = lds_off;                                              // lds_addr
    g0.z = (unsigned)(ga & 0xffffffffu);                         // global_addr[31:0]
    g0.w = (unsigned)((ga >> 32) & 0x01ffffffu) | 0x80000000u;   // global_addr[56:32] | type=2
    i32x8 g1;
    g1[0] = (int)0x20000u;                                       // data_size=4B, no flags
    g1[1] = (int)((strideDw & 0xffffu) << 16);                   // tensor_dim0[15:0]
    g1[2] = (int)((strideDw >> 16) | ((nRows & 0xffffu) << 16)); // tensor_dim0[31:16] | tensor_dim1[15:0]
    g1[3] = (int)((rowDw & 0xffffu) << 16);                      // tensor_dim1[31:16]=0 | tile_dim0
    g1[4] = (int)(nRows & 0xffffu);                              // tile_dim1 | tile_dim2=0
    g1[5] = (int)strideDw;                                       // tensor_dim0_stride[31:0]
    g1[6] = 0;                                                   // stride[47:32]=0 | dim1_stride lo
    g1[7] = 0;
    i32x4 gz = {0, 0, 0, 0};
#if defined(__clang_major__) && (__clang_major__ >= 23)
    i32x8 gz8 = {0, 0, 0, 0, 0, 0, 0, 0};
    __builtin_amdgcn_tensor_load_to_lds(g0, g1, gz, gz, gz8, 0);
#else
    __builtin_amdgcn_tensor_load_to_lds(g0, g1, gz, gz, 0);
#endif
}
#endif

// ---------- elementwise / setup kernels ----------

__global__ void k_transpose_in(const float* __restrict__ src, float* __restrict__ x, int L) {
    size_t i = (size_t)blockIdx.x * blockDim.x + threadIdx.x;
    if (i >= (size_t)L * D) return;
    int l = (int)(i / D), d = (int)(i % D);
    x[i] = src[(size_t)d * L + l];
}

__global__ void k_make_qbf(const float* __restrict__ x, const float* __restrict__ pos,
                           __bf16* __restrict__ qbf, int L) {
    size_t i = (size_t)blockIdx.x * blockDim.x + threadIdx.x;
    if (i >= (size_t)L * D) return;
    int l = (int)(i / D), d = (int)(i % D);
    qbf[i] = f2bf(x[i] + pos[(size_t)d * L + l]);
}

__global__ void k_f32_to_bf16(const float* __restrict__ in, __bf16* __restrict__ out, size_t n) {
    size_t i = (size_t)blockIdx.x * blockDim.x + threadIdx.x;
    if (i < n) out[i] = f2bf(in[i]);
}

// ---------- pack W[K,N] (f32) into WMMA B-fragment order (bf16), zero-padded ----
// Bp layout: [tile][k-step][lane][16 bf16]  (512 bf16 = 1KB per tile-step)
__global__ void k_pack_b(const float* __restrict__ Wm, __bf16* __restrict__ Bp,
                         int K, int N, int tilesPad)
{
    const int steps = K >> 5;
    const size_t total = (size_t)tilesPad * steps * 512;
    size_t i = (size_t)blockIdx.x * blockDim.x + threadIdx.x;
    if (i >= total) return;
    const int j    = (int)(i & 15);
    const int lane = (int)((i >> 4) & 31);
    const size_t rest = i >> 9;                 // tile*steps + s
    const int t = (int)(rest / steps);
    const int s = (int)(rest % steps);
    const int col = t * 16 + (lane & 15);
    const int k   = s * 32 + (lane >> 4) * 8 + (j >> 3) * 16 + (j & 7);
    Bp[i] = (col < N) ? f2bf(Wm[(size_t)k * N + col]) : f2bf(0.0f);
}

// ---------- WMMA GEMM ----------
// Block = 8 waves = 8 m-tiles x one group of 4 n-tiles; packed B staged in LDS
// (TDM tensor_load_to_lds when available) and shared by all waves.
__global__ void k_wmma_gemm(const __bf16* __restrict__ A, const __bf16* __restrict__ Bp,
                            const float* __restrict__ bias,
                            float* __restrict__ Cf, __bf16* __restrict__ Cbf,
                            int M, int N, int K, int nGroups, int relu)
{
    __shared__ __align__(128) char smem[4 * BCH * 1024];   // 32 KB

    const int lane = threadIdx.x;              // 0..31
    const int wave = threadIdx.y;              // 0..7
    const int tid  = wave * 32 + lane;
    const int tilesM = M >> 4;
    const int mb = blockIdx.x / nGroups;
    const int g  = blockIdx.x % nGroups;
    const int tm = mb * 8 + wave;
    const bool active = (tm < tilesM);         // wave-uniform
    const int hi = lane >> 4;
    const int ln = lane & 15;
    const int row = (active ? tm : 0) * 16 + ln;

    const int steps = K >> 5;
    const size_t tileStride = (size_t)steps * 512;          // bf16 per packed tile
    const __bf16* arow = A + (size_t)row * K + hi * 8;
    const char* gB = (const char*)(Bp + (size_t)(g * 4) * tileStride);

    v8f acc0 = {}, acc1 = {}, acc2 = {}, acc3 = {};

    for (int s0 = 0; s0 < steps; s0 += BCH) {
        const int ch = (steps - s0 < BCH) ? (steps - s0) : BCH;
        const unsigned rowBytes = (unsigned)ch * 1024u;
        const char* gsrc = gB + (size_t)s0 * 1024;

#if defined(USE_TDM)
        if (wave == 0) {
            tdm_load2d((unsigned)(size_t)(void*)smem, gsrc,
                       (unsigned)(ch * 256), 4u, (unsigned)(steps * 256));
            wait_tensor0();
        }
#else
        for (unsigned ofs = (unsigned)tid * 16u; ofs < 4u * rowBytes; ofs += 256u * 16u) {
            const unsigned u  = ofs / rowBytes;
            const unsigned in = ofs % rowBytes;
            *(int4*)(smem + (size_t)u * rowBytes + in) =
                *(const int4*)(gsrc + (size_t)u * (tileStride * 2) + in);
        }
#endif
        __syncthreads();

        if (active) {
            for (int sl = 0; sl < ch; ++sl) {
                const v8bf alo = *(const v8bf*)(arow);
                const v8bf ahi = *(const v8bf*)(arow + 16);
                const v16bf a = cat16(alo, ahi);
                const char* bb = smem + (size_t)sl * 1024 + (size_t)lane * 32;
                const v16bf b0 = *(const v16bf*)(bb);
                const v16bf b1 = *(const v16bf*)(bb + rowBytes);
                const v16bf b2 = *(const v16bf*)(bb + 2 * rowBytes);
                const v16bf b3 = *(const v16bf*)(bb + 3 * rowBytes);
                acc0 = __builtin_amdgcn_wmma_f32_16x16x32_bf16(false, a, false, b0, (short)0, acc0, false, false);
                acc1 = __builtin_amdgcn_wmma_f32_16x16x32_bf16(false, a, false, b1, (short)0, acc1, false, false);
                acc2 = __builtin_amdgcn_wmma_f32_16x16x32_bf16(false, a, false, b2, (short)0, acc2, false, false);
                acc3 = __builtin_amdgcn_wmma_f32_16x16x32_bf16(false, a, false, b3, (short)0, acc3, false, false);
                arow += 32;
            }
        }
        __syncthreads();
    }

    if (active) {
        v8f accs[4] = { acc0, acc1, acc2, acc3 };
        #pragma unroll
        for (int u = 0; u < 4; ++u) {
            const int col = (g * 4 + u) * 16 + ln;
            if (col < N) {
                const float bs = bias ? bias[col] : 0.0f;
                #pragma unroll
                for (int r = 0; r < 8; ++r) {
                    const int m = tm * 16 + hi * 8 + r;  // C: lanes 0-15 -> M=r, 16-31 -> M=8+r
                    float v = accs[u][r] + bs;
                    if (relu) v = fmaxf(v, 0.0f);
                    if (Cbf) Cbf[(size_t)m * N + col] = f2bf(v);
                    else     Cf [(size_t)m * N + col] = v;
                }
            }
        }
    }
}

// ---------- softmax over K2=25 per (l,h) row ----------
__global__ void k_softmax25(float* __restrict__ attn, int nrows) {
    int r = blockIdx.x * blockDim.x + threadIdx.x;
    if (r >= nrows) return;
    float* p = attn + (size_t)r * K2;
    float mx = p[0];
    #pragma unroll
    for (int k = 1; k < K2; ++k) mx = fmaxf(mx, p[k]);
    float s = 0.f;
    float e[K2];
    #pragma unroll
    for (int k = 0; k < K2; ++k) { e[k] = __expf(p[k] - mx); s += e[k]; }
    const float inv = 1.0f / s;
    #pragma unroll
    for (int k = 0; k < K2; ++k) p[k] = e[k] * inv;
}

// ---------- box-attention bilinear sampling; writes bf16 (feeds Wo GEMM) ----------
__global__ void k_box_sample(const float* __restrict__ val,   // L x D (f32)
                             const float* __restrict__ aw,    // L x NH*K2 (softmaxed)
                             const float* __restrict__ off,   // L x NH*4
                             __bf16* __restrict__ out,        // L x D (bf16)
                             int Hh, int Ww)
{
    const int l = blockIdx.x;
    const int t = threadIdx.x;     // 0..255 : channel = h*HD + d
    const int h = t >> 5;
    const int d = t & 31;
    const int wx = l % Ww, wy = l / Ww;
    const float cx = (wx + 0.5f) / (float)Ww;
    const float cy = (wy + 0.5f) / (float)Hh;

    const float* o = off + (size_t)l * (NH * 4) + h * 4;
    const float bx = cx + o[0] * (1.0f / 8.0f) * 0.025f;
    const float by = cy + o[1] * (1.0f / 8.0f) * 0.025f;
    const float bw = 0.025f + o[2] * (1.0f / 8.0f) * 0.025f;
    const float bh = 0.025f + o[3] * (1.0f / 8.0f) * 0.025f;
    const float* a = aw + (size_t)l * (NH * K2) + h * K2;

    float acc = 0.f;
    #pragma unroll 5
    for (int k = 0; k < K2; ++k) {
        const float gx = -0.5f + 0.25f * (float)(k % KK);
        const float gy = -0.5f + 0.25f * (float)(k / KK);
        const float px = (bx + bw * gx) * (float)Ww - 0.5f;
        const float py = (by + bh * gy) * (float)Hh - 0.5f;
        const float x0 = floorf(px), y0 = floorf(py);
        const float ak = a[k];
        #pragma unroll
        for (int c = 0; c < 4; ++c) {
            const float xi = x0 + (float)(c & 1);
            const float yi = y0 + (float)(c >> 1);
            const float wgt = (1.f - fabsf(px - xi)) * (1.f - fabsf(py - yi));
            const bool valid = (xi >= 0.f) && (xi < (float)Ww) && (yi >= 0.f) && (yi < (float)Hh);
            const int xc = min(max((int)xi, 0), Ww - 1);
            const int yc = min(max((int)yi, 0), Hh - 1);
            const float gval = val[(size_t)(yc * Ww + xc) * D + h * HD + d];
            acc += ak * (valid ? wgt : 0.f) * gval;
        }
    }
    out[(size_t)l * D + t] = f2bf(acc);
}

// ---------- residual add + LayerNorm (one 256-thread block per row) ----------
__global__ void k_add_ln(const float* __restrict__ xin, const float* __restrict__ res,
                         const float* __restrict__ g, const float* __restrict__ b,
                         float* __restrict__ dst)
{
    const int row = blockIdx.x;
    const int t = threadIdx.x;
    __shared__ float s[D];
    const float v = xin[(size_t)row * D + t] + res[(size_t)row * D + t];
    s[t] = v; __syncthreads();
    for (int st = D / 2; st > 0; st >>= 1) { if (t < st) s[t] += s[t + st]; __syncthreads(); }
    const float m = s[0] * (1.0f / D);
    __syncthreads();
    const float dv = v - m;
    s[t] = dv * dv; __syncthreads();
    for (int st = D / 2; st > 0; st >>= 1) { if (t < st) s[t] += s[t + st]; __syncthreads(); }
    const float var = s[0] * (1.0f / D);
    const float rin = rsqrtf(var + EPS);
    dst[(size_t)row * D + t] = dv * rin * g[t] + b[t];
}

// ---------- host ----------
static inline size_t alignup(size_t x) { return (x + 255) & ~(size_t)255; }

extern "C" void kernel_launch(void* const* d_in, const int* in_sizes, int n_in,
                              void* d_out, int out_size, void* d_ws, size_t ws_size,
                              hipStream_t stream)
{
    const float* src   = (const float*)d_in[0];
    const float* pos   = (const float*)d_in[1];
    const float* Wv    = (const float*)d_in[2];
    const float* bv    = (const float*)d_in[3];
    const float* Wbox  = (const float*)d_in[4];
    const float* bbox  = (const float*)d_in[5];
    const float* Wattn = (const float*)d_in[6];
    const float* battn = (const float*)d_in[7];
    const float* Wo    = (const float*)d_in[8];
    const float* bo    = (const float*)d_in[9];
    const float* W1    = (const float*)d_in[10];
    const float* b1    = (const float*)d_in[11];
    const float* W2    = (const float*)d_in[12];
    const float* b2    = (const float*)d_in[13];
    const float* ln1g  = (const float*)d_in[14];
    const float* ln1b  = (const float*)d_in[15];
    const float* ln2g  = (const float*)d_in[16];
    const float* ln2b  = (const float*)d_in[17];
    float* out = (float*)d_out;

    const int L = in_sizes[0] / D;               // H*W (B==1)
    int W = 1; while ((long long)W * W < L) ++W;  // H = W = sqrt(L)
    const int H = L / W;

    // workspace layout
    char* p = (char*)d_ws;
    float*  x     = (float*)p;  p += alignup((size_t)L * D * 4);
    __bf16* qbf   = (__bf16*)p; p += alignup((size_t)L * D * 2);
    __bf16* actbf = (__bf16*)p; p += alignup((size_t)L * D * 2);
    __bf16* ffbf  = (__bf16*)p; p += alignup((size_t)L * DFF * 2);
    float*  val   = (float*)p;  p += alignup((size_t)L * D * 4);
    float*  attn  = (float*)p;  p += alignup((size_t)L * NH * K2 * 4);
    float*  offb  = (float*)p;  p += alignup((size_t)L * NH * 4 * 4);
    float*  aout  = (float*)p;  p += alignup((size_t)L * D * 4);
    __bf16* wpk   = (__bf16*)p; p += alignup((size_t)2 * D * DFF * 2);  // packed B (padded)
    (void)ws_size;

    const size_t nLD = (size_t)L * D;
    const int eBlocks = (int)((nLD + 255) / 256);
    const dim3 gblk(32, 8);

    auto gemm = [&](const __bf16* A, const float* Wsrc, const float* bias,
                    float* Cf, __bf16* Cbf, int M, int N, int K, int relu) {
        const int tilesN   = (N + 15) >> 4;
        const int nGroups  = (tilesN + 3) >> 2;
        const int tilesPad = nGroups * 4;
        const int steps    = K >> 5;
        const size_t pkN   = (size_t)tilesPad * steps * 512;
        k_pack_b<<<(int)((pkN + 255) / 256), 256, 0, stream>>>(Wsrc, wpk, K, N, tilesPad);
        const int mBlocks = ((M >> 4) + 7) / 8;
        k_wmma_gemm<<<mBlocks * nGroups, gblk, 0, stream>>>(A, wpk, bias, Cf, Cbf,
                                                            M, N, K, nGroups, relu);
    };

    // x = transpose(src)
    k_transpose_in<<<eBlocks, 256, 0, stream>>>(src, x, L);

    for (int i = 0; i < NLAYER; ++i) {
        // q = x + pos (bf16), v-input = x (bf16)
        k_make_qbf<<<eBlocks, 256, 0, stream>>>(x, pos, qbf, L);
        k_f32_to_bf16<<<eBlocks, 256, 0, stream>>>(x, actbf, nLD);

        // projections (WMMA)
        gemm(actbf, Wv    + (size_t)i * D * D,       bv    + (size_t)i * D,       val,  nullptr, L, D,       D, 0);
        gemm(qbf,   Wattn + (size_t)i * D * NH * K2, battn + (size_t)i * NH * K2, attn, nullptr, L, NH * K2, D, 0);
        gemm(qbf,   Wbox  + (size_t)i * D * NH * 4,  bbox  + (size_t)i * NH * 4,  offb, nullptr, L, NH * 4,  D, 0);

        // softmax + bilinear sampling (bf16 out -> feeds Wo GEMM)
        k_softmax25<<<(L * NH + 255) / 256, 256, 0, stream>>>(attn, L * NH);
        k_box_sample<<<L, 256, 0, stream>>>(val, attn, offb, actbf, H, W);

        // output projection + residual LN1
        gemm(actbf, Wo + (size_t)i * D * D, bo + (size_t)i * D, aout, nullptr, L, D, D, 0);
        k_add_ln<<<L, D, 0, stream>>>(x, aout, ln1g + (size_t)i * D, ln1b + (size_t)i * D, x);

        // FFN: relu(x@W1+b1)@W2+b2, LN2
        k_f32_to_bf16<<<eBlocks, 256, 0, stream>>>(x, actbf, nLD);
        gemm(actbf, W1 + (size_t)i * D * DFF, b1 + (size_t)i * DFF, nullptr, ffbf, L, DFF, D, 1);
        gemm(ffbf,  W2 + (size_t)i * DFF * D, b2 + (size_t)i * D,   aout,  nullptr, L, D, DFF, 0);
        float* dst = (i == NLAYER - 1) ? out : x;
        k_add_ln<<<L, D, 0, stream>>>(x, aout, ln2g + (size_t)i * D, ln2b + (size_t)i * D, dst);
    }
    (void)n_in; (void)out_size;
}
